// CrossAttention_4973572129270
// MI455X (gfx1250) — matrix operand order repaired and checked by reading, not compiled, not run on hardware
//
#include <hip/hip_runtime.h>
#include <hip/hip_bf16.h>
#include <math.h>

typedef _Float16 v16h __attribute__((ext_vector_type(16)));
typedef _Float16 v8h  __attribute__((ext_vector_type(8)));
typedef float    v8f  __attribute__((ext_vector_type(8)));
typedef float    v4f  __attribute__((ext_vector_type(4)));

// ---------------------------------------------------------------------------
// WMMA fragment loaders (wave32, 16x16x32 f16, per CDNA5 ISA 7.12.2)
// A (16x32 MxK): lanes 0-15 row=lane, K={0..7,16..23}; lanes 16-31 row=lane-16, K={8..15,24..31}
// B (32x16 KxN): lanes 0-15 col=lane, K={0..7,16..23}; lanes 16-31 col=lane-16, K={8..15,24..31}
// ---------------------------------------------------------------------------
__device__ __forceinline__ v16h load_frag_a(const _Float16* base, int ldk) {
  const int lane = threadIdx.x & 31;
  const _Float16* p = base + (size_t)(lane & 15) * ldk + ((lane >> 4) ? 8 : 0);
  v8h lo = *reinterpret_cast<const v8h*>(p);
  v8h hi = *reinterpret_cast<const v8h*>(p + 16);
  v16h r;
#pragma unroll
  for (int i = 0; i < 8; ++i) { r[i] = lo[i]; r[8 + i] = hi[i]; }
  return r;
}

__device__ __forceinline__ v16h load_frag_b(const _Float16* base, int ldk) {
  const int lane = threadIdx.x & 31;
  const _Float16* p = base + (size_t)(lane & 15) * ldk + ((lane >> 4) ? 8 : 0);
  v8h lo = *reinterpret_cast<const v8h*>(p);
  v8h hi = *reinterpret_cast<const v8h*>(p + 16);
  v16h r;
#pragma unroll
  for (int i = 0; i < 8; ++i) { r[i] = lo[i]; r[8 + i] = hi[i]; }
  return r;
}

// ---------------------------------------------------------------------------
// Async DMA of one contiguous 8KB block (global -> LDS), 128 threads,
// 16B per lane per issue, 4 issues.  ISA 08: IOFFSET is added to BOTH the
// LDS address (vdst VGPR) and the memory address (saddr+vaddr), so the same
// vdst/vaddr pair covers all four 2KB slices.  Tracked on ASYNCcnt.
// ---------------------------------------------------------------------------
__device__ __forceinline__ void async_copy_8k(unsigned lds_base_byte,
                                              const _Float16* gsrc, int tid) {
  unsigned voff = (unsigned)tid * 16u;
  unsigned ldsa = lds_base_byte + voff;
  unsigned long long sbase = (unsigned long long)(uintptr_t)gsrc;
  asm volatile(
      "global_load_async_to_lds_b128 %0, %1, %2\n\t"
      "global_load_async_to_lds_b128 %0, %1, %2 offset:2048\n\t"
      "global_load_async_to_lds_b128 %0, %1, %2 offset:4096\n\t"
      "global_load_async_to_lds_b128 %0, %1, %2 offset:6144"
      :: "v"(ldsa), "v"(voff), "s"(sbase)
      : "memory");
}

// ---------------------------------------------------------------------------
// 1) Positional-encoding add, f32 -> f16 (raw-reshape semantics: within a
// batch, flat index f gets pe[f/128, f%128]).
// ---------------------------------------------------------------------------
__global__ void pe_add_f16(const float* __restrict__ in, _Float16* __restrict__ out,
                           int total, int perBatch) {
  const float kf = -9.21034037197618f / 128.0f; // -ln(10000)/dim
  for (int idx = blockIdx.x * blockDim.x + threadIdx.x; idx < total;
       idx += gridDim.x * blockDim.x) {
    int f = idx % perBatch;
    int n = f >> 7;
    int c = f & 127;
    float div = __expf(kf * (float)(c & ~1));
    float ang = (float)n * div;
    float pe = (c & 1) ? __cosf(ang) : __sinf(ang);
    out[idx] = (_Float16)(in[idx] + pe);
  }
}

// ---------------------------------------------------------------------------
// 2) Weight pre-swizzle: OIHW f32 (128,128,3,3) -> f16 [t=36][n=128][kk=32],
// t = tap*4 + cin/32 (implicit-GEMM k = tap*128 + cin).  Exactly the WMMA B
// operand layout -> hot loop is pure contiguous 16B fragment reads.
// ---------------------------------------------------------------------------
__global__ void wprep_f16(const float* __restrict__ w, _Float16* __restrict__ wh) {
  int idx = blockIdx.x * blockDim.x + threadIdx.x;
  if (idx >= 36 * 128 * 32) return;
  int kk = idx & 31;
  int n  = (idx >> 5) & 127;
  int t  = idx >> 12;
  int tap = t >> 2;
  int cin = ((t & 3) << 5) + kk;
  wh[idx] = (_Float16)w[(size_t)(n * 128 + cin) * 9 + tap];
}

// ---------------------------------------------------------------------------
// 3) Implicit-GEMM 3x3 conv (pad=1, 128->128ch) via WMMA f16 -> f32.
// Block = 128 threads (4 waves): 64 spatial rows x 128 out-ch.
// K loop: 36 steps of 32.  A-tile im2col'd to LDS; weight block (8KB/step)
// double-buffered into LDS by the async LDS-DMA, overlapped with the im2col
// fill and the previous step's WMMAs.  H,W compile-time -> div/mod become
// shifts.  Output written f16 NCHW flat == attention (row,128ch) layout.
// ---------------------------------------------------------------------------
template <int H, int W, bool RELU>
__global__ __launch_bounds__(128)
void conv3x3_wmma(const _Float16* __restrict__ in16, const _Float16* __restrict__ wh,
                  const float* __restrict__ bias, _Float16* __restrict__ out16) {
  constexpr int S = H * W;
  const int b = blockIdx.y;
  const int m0 = blockIdx.x * 64;
  const int tid = threadIdx.x;
  const int wave = tid >> 5;
  const int lane = tid & 31;

  __shared__ __align__(16) _Float16 Atile[64 * 32];
  __shared__ __align__(16) _Float16 Wtile[2 * 4096];

  const unsigned wt_base = (unsigned)(uintptr_t)(&Wtile[0]);

  v8f acc[8];
#pragma unroll
  for (int nt = 0; nt < 8; ++nt)
#pragma unroll
    for (int i = 0; i < 8; ++i) acc[nt][i] = 0.0f;

  const _Float16* inb = in16 + (size_t)b * 128 * S;

  // per-thread im2col invariants
  const int mm = tid & 63;
  const int kk0 = tid >> 6;          // 0 or 1
  const int m = m0 + mm;
  const int h0 = m / W;              // compile-time W -> shift
  const int w0 = m % W;

  // stage t=0 weights into buffer 0
  async_copy_8k(wt_base, wh, tid);

  for (int t = 0; t < 36; ++t) {
    const int tap = t >> 2;
    const int dh = tap / 3 - 1;
    const int dw = tap % 3 - 1;
    const int cinb = (t & 3) << 5;

    __syncthreads();  // everyone done with step t-1 (protects Atile + Wtile[(t+1)&1])

    if (t < 35)
      async_copy_8k(wt_base + (((t + 1) & 1) ? 8192u : 0u),
                    wh + (size_t)(t + 1) * 4096, tid);

    // im2col fill of A tile (overlaps with weight DMA in flight)
    {
      const int hh = h0 + dh;
      const int ww = w0 + dw;
      const bool ok = ((unsigned)hh < (unsigned)H) & ((unsigned)ww < (unsigned)W);
      const _Float16* src = inb + (size_t)(cinb + kk0) * S + hh * W + ww;
#pragma unroll
      for (int i = 0; i < 16; ++i) {
        _Float16 val = (_Float16)0.0f;
        if (ok) val = src[(size_t)(2 * i) * S];
        Atile[mm * 32 + kk0 + 2 * i] = val;
      }
    }

    // ensure step-t weight DMA (issued last iteration) has landed; the 4
    // just-issued step-(t+1) ops may stay in flight (async ops complete in order)
    if (t < 35) asm volatile("s_wait_asynccnt 0x4" ::: "memory");
    else        asm volatile("s_wait_asynccnt 0x0" ::: "memory");
    __syncthreads();

    v16h a = load_frag_a(&Atile[(wave * 16) * 32], 32);
    const _Float16* wl = &Wtile[(t & 1) * 4096];
    v16h bf[8];
#pragma unroll
    for (int nt = 0; nt < 8; ++nt) bf[nt] = load_frag_b(wl + nt * 512, 32);
#pragma unroll
    for (int nt = 0; nt < 8; ++nt)
      acc[nt] = __builtin_amdgcn_wmma_f32_16x16x32_f16(
          false, a, false, bf[nt], (short)0, acc[nt], false, false);
  }

  // Epilogue: D layout VGPR i -> row (lane/16)*8+i, col lane%16
  const int row0 = m0 + wave * 16 + ((lane >> 4) << 3);
#pragma unroll
  for (int nt = 0; nt < 8; ++nt) {
    int n = nt * 16 + (lane & 15);
    float bsv = bias[n];
    _Float16* ob = out16 + ((size_t)b * 128 + n) * S;
#pragma unroll
    for (int i = 0; i < 8; ++i) {
      float r = acc[nt][i] + bsv;
      if (RELU) r = fmaxf(r, 0.0f);
      ob[row0 + i] = (_Float16)r;
    }
  }
}

// ---------------------------------------------------------------------------
// 4) Flash-style attention, dv = 1.  One BLOCK per 16-query tile; the 4 waves
// split the 4096 keys (wave w takes chunks w, w+4, ...).  S^T = K_chunk.Q^T so
// each lane owns one query column; online softmax per lane, shfl_xor(16) to
// merge half-wave key groups, then LDS merge of the 4 wave partials.
// K fragments are register double-buffered across chunk iterations: the next
// chunk's 8 b128 loads issue as one clause before the current chunk's WMMA
// chain + softmax, hiding load latency and the WMMA->VALU hazard slots.
// ---------------------------------------------------------------------------
__global__ __launch_bounds__(128)
void attn_wmma(const _Float16* __restrict__ q16, const _Float16* __restrict__ k16,
               const float* __restrict__ z, float* __restrict__ out) {
  const int tid = threadIdx.x;
  const int wave = tid >> 5;
  const int lane = tid & 31;
  const int b = blockIdx.x >> 6;
  const int qb = blockIdx.x & 63;
  const float scale = 0.08838834764831845f; // 128^-0.5

  __shared__ float sm_m[4][16], sm_l[4][16], sm_a[4][16];

  const _Float16* qbase = q16 + ((size_t)b * 1024 + qb * 16) * 128;
  v16h bq[4];
#pragma unroll
  for (int ks = 0; ks < 4; ++ks) bq[ks] = load_frag_b(qbase + ks * 32, 128);

  const float* zb = z + (size_t)b * 4096;
  float m = -__builtin_inff(), l = 0.0f, av = 0.0f;

  const size_t kstride = (size_t)4 * 16 * 128;  // halves between this wave's chunks
  const _Float16* kb = k16 + ((size_t)b * 4096 + wave * 16) * 128;

  // prologue: fragments for this wave's first chunk
  v16h ac[4];
#pragma unroll
  for (int ks = 0; ks < 4; ++ks) ac[ks] = load_frag_a(kb + ks * 32, 128);

  for (int it = 0; it < 64; ++it) {
    const int kc = wave + it * 4;
    // next chunk (last iteration re-loads current: keeps loop straight-line,
    // wave-uniform, EXEC all-ones for the WMMAs)
    const _Float16* kb_next = (it < 63) ? (kb + kstride) : kb;
    __builtin_prefetch(kb + 2 * kstride + lane * 64, 0, 3); // L2 warm, 2 ahead

    v16h an[4];
#pragma unroll
    for (int ks = 0; ks < 4; ++ks) an[ks] = load_frag_a(kb_next + ks * 32, 128);

    v8f s;
#pragma unroll
    for (int i = 0; i < 8; ++i) s[i] = 0.0f;
#pragma unroll
    for (int ks = 0; ks < 4; ++ks)
      s = __builtin_amdgcn_wmma_f32_16x16x32_f16(
          false, ac[ks], false, bq[ks], (short)0, s, false, false);

    const float* vp = zb + kc * 16 + ((lane >> 4) << 3);
    v4f v0 = *reinterpret_cast<const v4f*>(vp);
    v4f v1 = *reinterpret_cast<const v4f*>(vp + 4);
    float vv[8];
#pragma unroll
    for (int i = 0; i < 4; ++i) { vv[i] = v0[i]; vv[4 + i] = v1[i]; }

    float sv[8], smax = -__builtin_inff();
#pragma unroll
    for (int i = 0; i < 8; ++i) { sv[i] = s[i] * scale; smax = fmaxf(smax, sv[i]); }
    float mn = fmaxf(m, smax);
    float corr = __expf(m - mn);
    l *= corr; av *= corr;
#pragma unroll
    for (int i = 0; i < 8; ++i) {
      float e = __expf(sv[i] - mn);
      l += e;
      av += e * vv[i];
    }
    m = mn;

#pragma unroll
    for (int ks = 0; ks < 4; ++ks) ac[ks] = an[ks];
    kb = kb_next;
  }

  // merge the two half-wave key groups (same query column)
  float m2 = __shfl_xor(m, 16, 32);
  float l2 = __shfl_xor(l, 16, 32);
  float a2 = __shfl_xor(av, 16, 32);
  float M = fmaxf(m, m2);
  float c1 = __expf(m - M), c2 = __expf(m2 - M);
  float Lw = l * c1 + l2 * c2;
  float Aw = av * c1 + a2 * c2;

  if (lane < 16) { sm_m[wave][lane] = M; sm_l[wave][lane] = Lw; sm_a[wave][lane] = Aw; }
  __syncthreads();

  if (tid < 16) {
    float Mg = sm_m[0][tid];
#pragma unroll
    for (int w = 1; w < 4; ++w) Mg = fmaxf(Mg, sm_m[w][tid]);
    float L = 0.0f, A = 0.0f;
#pragma unroll
    for (int w = 0; w < 4; ++w) {
      float c = __expf(sm_m[w][tid] - Mg);
      L += sm_l[w][tid] * c;
      A += sm_a[w][tid] * c;
    }
    out[(size_t)b * 1024 + qb * 16 + tid] = A / L;
  }
}

// ---------------------------------------------------------------------------
extern "C" void kernel_launch(void* const* d_in, const int* in_sizes, int n_in,
                              void* d_out, int out_size, void* d_ws, size_t ws_size,
                              hipStream_t stream) {
  (void)in_sizes; (void)n_in; (void)out_size; (void)ws_size;
  const float* x   = (const float*)d_in[0];
  const float* y   = (const float*)d_in[1];
  const float* z   = (const float*)d_in[2];
  const float* wq1 = (const float*)d_in[3];
  const float* bq1 = (const float*)d_in[4];
  const float* wq2 = (const float*)d_in[5];
  const float* bq2 = (const float*)d_in[6];
  const float* wk1 = (const float*)d_in[7];
  const float* bk1 = (const float*)d_in[8];
  const float* wk2 = (const float*)d_in[9];
  const float* bk2 = (const float*)d_in[10];
  float* out = (float*)d_out;

  char* ws = (char*)d_ws;
  size_t off = 0;
  auto alloc = [&](size_t bytes) {
    char* p = ws + off;
    off += (bytes + 255) & ~(size_t)255;
    return p;
  };
  _Float16* x16  = (_Float16*)alloc(8ull * 128 * 1024 * 2);
  _Float16* y16  = (_Float16*)alloc(8ull * 128 * 4096 * 2);
  _Float16* t1x  = (_Float16*)alloc(8ull * 128 * 1024 * 2);
  _Float16* q16  = (_Float16*)alloc(8ull * 128 * 1024 * 2);
  _Float16* t1y  = (_Float16*)alloc(8ull * 128 * 4096 * 2);
  _Float16* k16  = (_Float16*)alloc(8ull * 128 * 4096 * 2);
  _Float16* wq1h = (_Float16*)alloc(36ull * 128 * 32 * 2);
  _Float16* wq2h = (_Float16*)alloc(36ull * 128 * 32 * 2);
  _Float16* wk1h = (_Float16*)alloc(36ull * 128 * 32 * 2);
  _Float16* wk2h = (_Float16*)alloc(36ull * 128 * 32 * 2);

  // PE add (f32 -> f16)
  pe_add_f16<<<2048, 256, 0, stream>>>(x, x16, 8 * 128 * 1024, 128 * 1024);
  pe_add_f16<<<2048, 256, 0, stream>>>(y, y16, 8 * 128 * 4096, 128 * 4096);

  // Weight swizzle to WMMA B layout
  const int wN = 36 * 128 * 32;
  wprep_f16<<<(wN + 255) / 256, 256, 0, stream>>>(wq1, wq1h);
  wprep_f16<<<(wN + 255) / 256, 256, 0, stream>>>(wq2, wq2h);
  wprep_f16<<<(wN + 255) / 256, 256, 0, stream>>>(wk1, wk1h);
  wprep_f16<<<(wN + 255) / 256, 256, 0, stream>>>(wk2, wk2h);

  // q path: 32x32, 16 spatial tiles x 8 batches
  conv3x3_wmma<32, 32, true ><<<dim3(16, 8), 128, 0, stream>>>(x16, wq1h, bq1, t1x);
  conv3x3_wmma<32, 32, false><<<dim3(16, 8), 128, 0, stream>>>(t1x, wq2h, bq2, q16);
  // k path: 64x64, 64 tiles x 8 batches
  conv3x3_wmma<64, 64, true ><<<dim3(64, 8), 128, 0, stream>>>(y16, wk1h, bk1, t1y);
  conv3x3_wmma<64, 64, false><<<dim3(64, 8), 128, 0, stream>>>(t1y, wk2h, bk2, k16);

  // attention: one block per (batch, 16-query tile); 4 waves split the keys
  attn_wmma<<<512, 128, 0, stream>>>(q16, k16, z, out);
}
